// MultiVariateAttention_20263655702738
// MI455X (gfx1250) — compile-verified, matrix operand order checked by reading, not run on hardware
//
#include <hip/hip_runtime.h>
#include <hip/hip_bf16.h>

// MVN-diagonal log-density -> activation map, refactored as a K=32 f16 WMMA GEMM.
//
//   log2p[b,u] = sum_k Apack[b,k] * Bpack[u,k] - Cterm[u]     (base-2 domain)
//   Apack[b, 0:16]  = x_d^2          Bpack[u, 0:16]  = log2e * (-0.5 / s_d^2)
//   Apack[b,16:32]  = x_d            Bpack[u,16:32]  = log2e * ( mu_d / s_d^2)
//   Cterm[u] = log2e * (sum_d log s_d + 0.5*sum_d mu_d^2/s_d^2 + D*0.5*log 2pi)
//   out = exp2(log2p)                                          (v_exp_f32)
//
// One V_WMMA_F32_16X16X32_F16 computes a full 16x16 output tile: the K=32
// shape covers the entire D=16 reduction (quadratic + linear terms) in a
// single matrix op. Kernel is bound by the 64MB output stream (~2.8us at
// 23.3 TB/s); params (528KB) stay L2/WGP$-resident.

typedef __attribute__((ext_vector_type(16))) _Float16 v16h;
typedef __attribute__((ext_vector_type(8)))  _Float16 v8h;
typedef __attribute__((ext_vector_type(8)))  float    v8f;

#define D_DIMS 16
#define LOG2E  1.44269504088896340736f

__global__ void mvn_prep_units(const float* __restrict__ units,
                               const float* __restrict__ attn,
                               _Float16* __restrict__ Bpack,
                               float* __restrict__ Cterm,
                               int U) {
  int u = blockIdx.x * blockDim.x + threadIdx.x;
  if (u >= U) return;
  const float* arow = attn  + (size_t)u * D_DIMS * D_DIMS;
  const float* urow = units + (size_t)u * D_DIMS;
  float cacc = (float)D_DIMS * 0.91893853320467274178f; // D * 0.5*log(2*pi)
#pragma unroll
  for (int d = 0; d < D_DIMS; ++d) {
    float s  = fmaxf(arow[d * D_DIMS + d], 1e-6f);  // diagonal of scale_tril
    float w  = 1.0f / (s * s);
    float mu = urow[d];
    Bpack[(size_t)u * 32 + d]      = (_Float16)(LOG2E * -0.5f * w);
    Bpack[(size_t)u * 32 + 16 + d] = (_Float16)(LOG2E * mu * w);
    cacc += __logf(s) + 0.5f * mu * mu * w;
  }
  Cterm[u] = LOG2E * cacc;
}

__global__ void mvn_prep_x(const float* __restrict__ inp,
                           _Float16* __restrict__ Apack,
                           int B) {
  int b = blockIdx.x * blockDim.x + threadIdx.x;
  if (b >= B) return;
  const float* xrow = inp + (size_t)b * D_DIMS;
#pragma unroll
  for (int d = 0; d < D_DIMS; ++d) {
    float x = xrow[d];
    Apack[(size_t)b * 32 + d]      = (_Float16)(x * x);
    Apack[(size_t)b * 32 + 16 + d] = (_Float16)x;
  }
}

// One wave (32 lanes) owns a contiguous stripe of U-tiles; the A fragment
// (16 stimuli) is loaded once and reused across the whole stripe. All inner
// addressing is pointer-bump + immediate offsets when UC (columns) is a
// compile-time constant.
//
// f16 A-fragment lane layout (CDNA5 ISA, 16-bit A 16x32):
//   lane L: row M = L&15, half = L>>4
//   v16h elements 0..7  hold K = 8*half + 0..7
//   v16h elements 8..15 hold K = 16 + 8*half + 0..7
// B fragment packed symmetrically with column N = L&15.
// C/D layout: VGPR g, lane -> row M = g + 8*half, col N = L&15.
template <int UC>  // UC > 0: compile-time column count; UC == 0: runtime
__global__ void __launch_bounds__(256)
mvn_wmma_exp(const _Float16* __restrict__ Apack,
             const _Float16* __restrict__ Bpack,
             const float* __restrict__ Cterm,
             float* __restrict__ out,
             int Urt) {
  const int U    = (UC > 0) ? UC : Urt;
  const int lane = threadIdx.x & 31;
  const int wave = threadIdx.x >> 5;   // 0..7
  const int half = lane >> 4;          // 0 or 1
  const int mn   = lane & 15;

  const int mBase = blockIdx.x * 16;   // 16-stimulus stripe

  // ---- A fragment: two contiguous 16B loads per lane, reused for all tiles
  const _Float16* Ap = Apack + ((size_t)(mBase + mn)) * 32 + half * 8;
  v8h alo = *(const v8h*)(Ap);
  v8h ahi = *(const v8h*)(Ap + 16);
  v16h a;
#pragma unroll
  for (int e = 0; e < 8; ++e) { a[e] = alo[e]; a[e + 8] = ahi[e]; }

  // ---- contiguous stripe of U-tiles for this wave
  const int tilesPerWave = (U >> 4) >> 3;          // (U/16)/8 waves
  const int ut0  = wave * tilesPerWave;
  const int col0 = ut0 * 16 + mn;

  const _Float16* Bp = Bpack + (size_t)col0 * 32 + half * 8;
  const float*    Cp = Cterm + col0;
  float*          op = out + (size_t)(mBase + 8 * half) * U + col0;

#pragma unroll 2
  for (int t = 0; t < tilesPerWave; ++t) {
    // speculative prefetch 4 tiles ahead (global_prefetch_b8; OOB is dropped)
    __builtin_prefetch(Bp + 4 * 16 * 32, 0, 1);

    v8h blo = *(const v8h*)(Bp);
    v8h bhi = *(const v8h*)(Bp + 16);
    v16h bfrag;
#pragma unroll
    for (int e = 0; e < 8; ++e) { bfrag[e] = blo[e]; bfrag[e + 8] = bhi[e]; }

    v8f acc = {};
    // D = A(16x32) x B(32x16) + 0  ->  v_wmma_f32_16x16x32_f16
    acc = __builtin_amdgcn_wmma_f32_16x16x32_f16(
        /*neg_a=*/false, a, /*neg_b=*/false, bfrag,
        /*c_mod=*/(short)0, acc, /*reuse_a=*/false, /*reuse_b=*/false);

    const float ct = *Cp;
#pragma unroll
    for (int g = 0; g < 8; ++g) {
      // exp2 domain: one sub + v_exp_f32 per element; when UC is a constant,
      // g*U folds into the store's immediate offset field.
      op[(size_t)g * U] = __builtin_amdgcn_exp2f(acc[g] - ct);
    }

    Bp += 16 * 32;   // next tile's 16 columns
    Cp += 16;
    op += 16;
  }
}

extern "C" void kernel_launch(void* const* d_in, const int* in_sizes, int n_in,
                              void* d_out, int out_size, void* d_ws, size_t ws_size,
                              hipStream_t stream) {
  const float* inp   = (const float*)d_in[0];  // [B,1,D] f32
  const float* units = (const float*)d_in[1];  // [U,D]   f32
  const float* attn  = (const float*)d_in[2];  // [U,D,D] f32
  float* out = (float*)d_out;                  // [B,U]   f32

  const int B = in_sizes[0] / D_DIMS;                     // 4096
  const int U = in_sizes[2] / (D_DIMS * D_DIMS);          // 4096

  // workspace: Bpack (U*32 f16) | Apack (B*32 f16) | Cterm (U f32)
  char* ws = (char*)d_ws;
  _Float16* Bpack = (_Float16*)ws;
  _Float16* Apack = (_Float16*)(ws + (size_t)U * 32 * sizeof(_Float16));
  float*    Cterm = (float*)   (ws + (size_t)U * 32 * sizeof(_Float16)
                                   + (size_t)B * 32 * sizeof(_Float16));

  mvn_prep_units<<<(U + 255) / 256, 256, 0, stream>>>(units, attn, Bpack, Cterm, U);
  mvn_prep_x   <<<(B + 255) / 256, 256, 0, stream>>>(inp, Apack, B);

  if (U == 4096) {
    mvn_wmma_exp<4096><<<B / 16, 256, 0, stream>>>(Apack, Bpack, Cterm, out, U);
  } else {
    mvn_wmma_exp<0>   <<<B / 16, 256, 0, stream>>>(Apack, Bpack, Cterm, out, U);
  }
}
